// CrossAttention_27376121545454
// MI455X (gfx1250) — compile-verified, matrix operand order checked by reading, not run on hardware
//
#include <hip/hip_runtime.h>
#include <hip/hip_bf16.h>
#include <math.h>

// ---------------------------------------------------------------------------
// CrossAttention for MI455X (gfx1250, wave32, WMMA 16x16x32 bf16)
// ---------------------------------------------------------------------------

typedef __bf16 bf16_t;
typedef __bf16 v16bf __attribute__((ext_vector_type(16)));
typedef __bf16 v8bf  __attribute__((ext_vector_type(8)));
typedef __bf16 v4bf  __attribute__((ext_vector_type(4)));
typedef float  v8f   __attribute__((ext_vector_type(8)));

#define BATCH 4
#define NQ    4096
#define NM    1024
#define QDIM  1024
#define CDIM  768
#define HEADS 16
#define DH    64
#define INNER 1024

__device__ __forceinline__ v8f wmma_bf16(v16bf a, v16bf b, v8f c) {
  // (neg_a, A, neg_b, B, c_mod, C, reuse_a, reuse_b)
  return __builtin_amdgcn_wmma_f32_16x16x32_bf16(false, a, false, b, (short)0, c,
                                                 false, false);
}

__device__ __forceinline__ v16bf cat16(v8bf lo, v8bf hi) {
  v16bf r;
#pragma unroll
  for (int i = 0; i < 8; ++i) { r[i] = lo[i]; r[i + 8] = hi[i]; }
  return r;
}

// Load a 16x32 WMMA operand (A or B side) from K-contiguous bf16 storage.
// base -> element [tile_origin][k0]; stride = elements per row (mult of 8).
// wave32 layout: row/col = lane&15, half = lane>>4,
//   elems 0-7  : k = 8*half .. 8*half+7
//   elems 8-15 : k = 16+8*half .. 16+8*half+7
__device__ __forceinline__ v16bf load_op(const bf16_t* base, int stride, int lane) {
  const int h = (lane >> 4) & 1;
  const int r = lane & 15;
  const bf16_t* p = base + (size_t)r * stride + 8 * h;
  v8bf lo = *(const v8bf*)(p);
  v8bf hi = *(const v8bf*)(p + 16);
  return cat16(lo, hi);
}

// Software-pipelined 16x64 GEMM tile: acc[t] += A[16xK] * Bt[t][16xK]^T.
// Double-buffers all operands in registers: next k-step's 10 b128 loads are
// in flight while the current step's 4 WMMAs issue (partial loadcnt waits).
__device__ __forceinline__ void gemm16x64(const bf16_t* __restrict__ a,
                                          const bf16_t* __restrict__ bt,
                                          int K, int lane, v8f acc[4]) {
  const int prow = (size_t)(lane & 15) * K;  // per-lane row for prefetch
  v16bf ac = load_op(a, K, lane);
  v16bf bc[4];
#pragma unroll
  for (int t = 0; t < 4; ++t) bc[t] = load_op(bt + (size_t)t * 16 * K, K, lane);
  for (int k = 32; k < K; k += 32) {
    v16bf an = load_op(a + k, K, lane);
    v16bf bn[4];
#pragma unroll
    for (int t = 0; t < 4; ++t)
      bn[t] = load_op(bt + (size_t)t * 16 * K + k, K, lane);
    __builtin_prefetch(a + prow + k + 64, 0, 1);   // warm L2 two steps ahead
#pragma unroll
    for (int t = 0; t < 4; ++t) acc[t] = wmma_bf16(ac, bc[t], acc[t]);
    ac = an;
#pragma unroll
    for (int t = 0; t < 4; ++t) bc[t] = bn[t];
  }
#pragma unroll
  for (int t = 0; t < 4; ++t) acc[t] = wmma_bf16(ac, bc[t], acc[t]);
}

// --------------------------- conversion kernels ----------------------------

__global__ void cvt_bf16_kernel(const float* __restrict__ src,
                                bf16_t* __restrict__ dst, int n) {
  int i = (blockIdx.x * 256 + threadIdx.x) * 4;
  if (i + 3 < n) {
    float4 v = *(const float4*)(src + i);
    v4bf o;
    o[0] = (bf16_t)v.x; o[1] = (bf16_t)v.y; o[2] = (bf16_t)v.z; o[3] = (bf16_t)v.w;
    *(v4bf*)(dst + i) = o;
  }
}

// src [R][C] fp32 -> dst [C][R] bf16 (weights become [out][in]: K-contiguous)
__global__ void cvt_t_bf16_kernel(const float* __restrict__ src,
                                  bf16_t* __restrict__ dst, int R, int C) {
  int idx = blockIdx.x * 256 + threadIdx.x;
  if (idx < R * C) {
    int r = idx / C, c = idx - r * C;
    dst[(size_t)c * R + r] = (bf16_t)src[idx];
  }
}

// ------------------------------ projections --------------------------------
// C[row][col] = scale * sum_k A[row][k] * Bt[col][k]; one wave = 16x64 tile.

__global__ void proj_bf16_kernel(const bf16_t* __restrict__ A,
                                 const bf16_t* __restrict__ Bt,
                                 bf16_t* __restrict__ C,
                                 int K, int ldc, float scale) {
  const int lane = threadIdx.x & 31;
  const int wave = threadIdx.x >> 5;
  const int row0 = (blockIdx.y * 8 + wave) * 16;
  const int col0 = blockIdx.x * 64;
  v8f acc[4] = {};
  gemm16x64(A + (size_t)row0 * K, Bt + (size_t)col0 * K, K, lane, acc);
  const int h = lane >> 4, c = lane & 15;
#pragma unroll
  for (int t = 0; t < 4; ++t)
#pragma unroll
    for (int e = 0; e < 8; ++e)
      C[(size_t)(row0 + e + 8 * h) * ldc + col0 + t * 16 + c] =
          (bf16_t)(acc[t][e] * scale);
}

// V projection, output transposed per head: Vt[b][h][d][m] (m contiguous),
// so that P@V's B-operand is K(=m)-contiguous. One x-block = one head.
__global__ void projv_bf16_kernel(const bf16_t* __restrict__ A,
                                  const bf16_t* __restrict__ Bt,
                                  bf16_t* __restrict__ Vt, int K) {
  const int lane = threadIdx.x & 31;
  const int wave = threadIdx.x >> 5;
  const int row0 = (blockIdx.y * 8 + wave) * 16;   // global ctx row: b*NM + m
  const int head = blockIdx.x;
  v8f acc[4] = {};
  gemm16x64(A + (size_t)row0 * K, Bt + (size_t)(head * DH) * K, K, lane, acc);
  const int h = lane >> 4, c = lane & 15;
  const int mg = row0 + 8 * h;        // consecutive accumulator elems -> m+1
  const int b  = mg >> 10;            // NM == 1024
  const int m0 = mg & (NM - 1);
#pragma unroll
  for (int t = 0; t < 4; ++t) {
    v8bf pk;
#pragma unroll
    for (int e = 0; e < 8; ++e) pk[e] = (bf16_t)acc[t][e];
    size_t off = (((size_t)b * HEADS + head) * DH + t * 16 + c) * NM + m0;
    *(v8bf*)(Vt + off) = pk;          // free transpose: 16B contiguous store
  }
}

// --------------------------- flash attention -------------------------------
// grid: x = N/64 query tiles, y = B*H. Block 128 = 4 waves; wave = 16 q rows.
// Q pre-scaled by 1/sqrt(DH). Row stats live per accumulator element
// (row = elem + 8*(lane>>4)), uniform across the 16 lanes of each half-wave.

__global__ void fa_kernel(const bf16_t* __restrict__ Q,
                          const bf16_t* __restrict__ Kc,
                          const bf16_t* __restrict__ Vt,
                          bf16_t* __restrict__ O) {
  __shared__ __align__(16) bf16_t Plds[4][16][72];  // 144B rows: conflict-free
  const int lane = threadIdx.x & 31;
  const int wave = threadIdx.x >> 5;
  const int b  = blockIdx.y >> 4;
  const int hd = blockIdx.y & 15;
  const int n0 = blockIdx.x * 64 + wave * 16;
  const int h = lane >> 4, c = lane & 15;

  const bf16_t* qbase = Q + ((size_t)(b * NQ + n0)) * INNER + hd * DH;
  v16bf Aq0 = load_op(qbase,      INNER, lane);   // d = 0..31
  v16bf Aq1 = load_op(qbase + 32, INNER, lane);   // d = 32..63

  const bf16_t* kbB = Kc + ((size_t)b * NM) * INNER + hd * DH;
  const bf16_t* vbB = Vt + (((size_t)b * HEADS + hd) * DH) * NM;

  v8f Oacc[4] = {};
  float mrow[8], lrow[8];
#pragma unroll
  for (int e = 0; e < 8; ++e) { mrow[e] = -1e30f; lrow[e] = 0.0f; }

  for (int mt = 0; mt < NM; mt += 64) {
    const bf16_t* kb0 = kbB + (size_t)mt * INNER;
    const bf16_t* vb0 = vbB + mt;

    // warm L2 for the next M-tile while this one computes
    if (mt + 64 < NM) {
      __builtin_prefetch(kb0 + (size_t)(64 + c) * INNER, 0, 1);
      __builtin_prefetch(vb0 + (size_t)c * NM + 64, 0, 1);
    }

    // ---- batch-issue all 8 K operands, then S = Q K^T (16x64)
    v16bf Bk[8];
#pragma unroll
    for (int jt = 0; jt < 4; ++jt) {
      Bk[2 * jt]     = load_op(kb0 + (size_t)jt * 16 * INNER,      INNER, lane);
      Bk[2 * jt + 1] = load_op(kb0 + (size_t)jt * 16 * INNER + 32, INNER, lane);
    }
    v8f S[4];
#pragma unroll
    for (int jt = 0; jt < 4; ++jt) {
      v8f z = {};
      S[jt] = wmma_bf16(Aq1, Bk[2 * jt + 1], wmma_bf16(Aq0, Bk[2 * jt], z));
    }

    // ---- issue all 8 V operands now; softmax math below hides their latency
    v16bf Bv[8];
#pragma unroll
    for (int dt = 0; dt < 4; ++dt) {
      Bv[2 * dt]     = load_op(vb0 + (size_t)dt * 16 * NM,      NM, lane);
      Bv[2 * dt + 1] = load_op(vb0 + (size_t)dt * 16 * NM + 32, NM, lane);
    }

    // ---- row max over the 64 keys of this tile
    v8f ml;
#pragma unroll
    for (int e = 0; e < 8; ++e)
      ml[e] = fmaxf(fmaxf(S[0][e], S[1][e]), fmaxf(S[2][e], S[3][e]));
#pragma unroll
    for (int mask = 1; mask <= 8; mask <<= 1)
#pragma unroll
      for (int e = 0; e < 8; ++e)
        ml[e] = fmaxf(ml[e], __shfl_xor(ml[e], mask, 32));

    // ---- online rescale
    v8f corr;
#pragma unroll
    for (int e = 0; e < 8; ++e) {
      float mnew = fmaxf(mrow[e], ml[e]);
      corr[e] = __expf(mrow[e] - mnew);
      mrow[e] = mnew;
      lrow[e] *= corr[e];
    }
#pragma unroll
    for (int t = 0; t < 4; ++t)
#pragma unroll
      for (int e = 0; e < 8; ++e) Oacc[t][e] *= corr[e];

    // ---- P = exp(S - m), stage to LDS (C-layout -> A-layout via LDS)
    __syncthreads();
    v8f ps = {};
#pragma unroll
    for (int t = 0; t < 4; ++t)
#pragma unroll
      for (int e = 0; e < 8; ++e) {
        float p = __expf(S[t][e] - mrow[e]);
        ps[e] += p;
        Plds[wave][e + 8 * h][t * 16 + c] = (bf16_t)p;
      }
#pragma unroll
    for (int mask = 1; mask <= 8; mask <<= 1)
#pragma unroll
      for (int e = 0; e < 8; ++e) ps[e] += __shfl_xor(ps[e], mask, 32);
#pragma unroll
    for (int e = 0; e < 8; ++e) lrow[e] += ps[e];
    __syncthreads();

    // ---- O += P @ V  (V operands already resident in registers)
    v16bf Ap0 = load_op(&Plds[wave][0][0],  72, lane);
    v16bf Ap1 = load_op(&Plds[wave][0][32], 72, lane);
#pragma unroll
    for (int dt = 0; dt < 4; ++dt) {
      Oacc[dt] = wmma_bf16(Ap0, Bv[2 * dt],     Oacc[dt]);
      Oacc[dt] = wmma_bf16(Ap1, Bv[2 * dt + 1], Oacc[dt]);
    }
  }

  // ---- normalize and store bf16 attention output [B*N][INNER]
  v8f inv;
#pragma unroll
  for (int e = 0; e < 8; ++e) inv[e] = 1.0f / lrow[e];
#pragma unroll
  for (int dt = 0; dt < 4; ++dt)
#pragma unroll
    for (int e = 0; e < 8; ++e)
      O[((size_t)(b * NQ + n0 + e + 8 * h)) * INNER + hd * DH + dt * 16 + c] =
          (bf16_t)(Oacc[dt][e] * inv[e]);
}

// ---------------------------- output projection ----------------------------

__global__ void outproj_kernel(const bf16_t* __restrict__ A,
                               const bf16_t* __restrict__ Bt,
                               const float* __restrict__ bias,
                               float* __restrict__ Cout, int K) {
  const int lane = threadIdx.x & 31;
  const int wave = threadIdx.x >> 5;
  const int row0 = (blockIdx.y * 8 + wave) * 16;
  const int col0 = blockIdx.x * 64;
  v8f acc[4] = {};
  gemm16x64(A + (size_t)row0 * K, Bt + (size_t)col0 * K, K, lane, acc);
  const int h = lane >> 4, c = lane & 15;
#pragma unroll
  for (int t = 0; t < 4; ++t) {
    float bb = bias[col0 + t * 16 + c];
#pragma unroll
    for (int e = 0; e < 8; ++e)
      Cout[(size_t)(row0 + e + 8 * h) * QDIM + col0 + t * 16 + c] = acc[t][e] + bb;
  }
}

// ------------------------------- launcher ----------------------------------

extern "C" void kernel_launch(void* const* d_in, const int* in_sizes, int n_in,
                              void* d_out, int out_size, void* d_ws, size_t ws_size,
                              hipStream_t stream) {
  (void)in_sizes; (void)n_in; (void)out_size; (void)ws_size;
  const float* x   = (const float*)d_in[0];
  const float* ctx = (const float*)d_in[1];
  const float* Wq  = (const float*)d_in[2];
  const float* Wk  = (const float*)d_in[3];
  const float* Wv  = (const float*)d_in[4];
  const float* Wo  = (const float*)d_in[5];
  const float* bo  = (const float*)d_in[6];
  float* out = (float*)d_out;

  char* ws = (char*)d_ws;
  size_t cur = 0;
  auto carve = [&](size_t bytes) -> void* {
    void* p = ws + cur;
    cur = (cur + bytes + 255) & ~(size_t)255;
    return p;
  };
  bf16_t* xb  = (bf16_t*)carve((size_t)BATCH * NQ * QDIM * 2);   // x bf16
  bf16_t* cb  = (bf16_t*)carve((size_t)BATCH * NM * CDIM * 2);   // ctx bf16
  bf16_t* Wqt = (bf16_t*)carve((size_t)QDIM * INNER * 2);        // [out][in]
  bf16_t* Wkt = (bf16_t*)carve((size_t)CDIM * INNER * 2);
  bf16_t* Wvt = (bf16_t*)carve((size_t)CDIM * INNER * 2);
  bf16_t* Wot = (bf16_t*)carve((size_t)INNER * QDIM * 2);
  bf16_t* Qb  = (bf16_t*)carve((size_t)BATCH * NQ * INNER * 2);  // pre-scaled
  bf16_t* Kb  = (bf16_t*)carve((size_t)BATCH * NM * INNER * 2);
  bf16_t* Vt  = (bf16_t*)carve((size_t)BATCH * NM * INNER * 2);  // [B][H][DH][M]
  bf16_t* Ob  = (bf16_t*)carve((size_t)BATCH * NQ * INNER * 2);  // attn out

  const int nx = BATCH * NQ * QDIM;
  const int nc = BATCH * NM * CDIM;
  cvt_bf16_kernel<<<nx / 1024, 256, 0, stream>>>(x, xb, nx);
  cvt_bf16_kernel<<<nc / 1024, 256, 0, stream>>>(ctx, cb, nc);
  cvt_t_bf16_kernel<<<(QDIM * INNER + 255) / 256, 256, 0, stream>>>(Wq, Wqt, QDIM, INNER);
  cvt_t_bf16_kernel<<<(CDIM * INNER + 255) / 256, 256, 0, stream>>>(Wk, Wkt, CDIM, INNER);
  cvt_t_bf16_kernel<<<(CDIM * INNER + 255) / 256, 256, 0, stream>>>(Wv, Wvt, CDIM, INNER);
  cvt_t_bf16_kernel<<<(INNER * QDIM + 255) / 256, 256, 0, stream>>>(Wo, Wot, INNER, QDIM);

  // Q = x @ Wq, scale 1/sqrt(DH)=0.125 folded in (free)
  proj_bf16_kernel<<<dim3(INNER / 64, BATCH * NQ / 128), 256, 0, stream>>>(
      xb, Wqt, Qb, QDIM, INNER, 0.125f);
  // K = ctx @ Wk
  proj_bf16_kernel<<<dim3(INNER / 64, BATCH * NM / 128), 256, 0, stream>>>(
      cb, Wkt, Kb, CDIM, INNER, 1.0f);
  // V = ctx @ Wv, stored transposed per head
  projv_bf16_kernel<<<dim3(HEADS, BATCH * NM / 128), 256, 0, stream>>>(
      cb, Wvt, Vt, CDIM);
  // fused softmax(QK^T)V
  fa_kernel<<<dim3(NQ / 64, BATCH * HEADS), 128, 0, stream>>>(Qb, Kb, Vt, Ob);
  // out = attn @ Wo + bo  (fp32)
  outproj_kernel<<<dim3(QDIM / 64, BATCH * NQ / 128), 256, 0, stream>>>(
      Ob, Wot, bo, out, INNER);
}
